// FourierCrossAttention_11458972745768
// MI455X (gfx1250) — matrix operand order repaired and checked by reading, not compiled, not run on hardware
//
#include <hip/hip_runtime.h>
#include <hip/hip_bf16.h>
#include <math.h>

// Problem constants (from reference)
#define BB 16
#define EE 64
#define HH 8
#define LL 2048
#define MODES 64
#define NC 128          // 2*MODES (re/im interleaved)
#define MROWS 8192      // B*H*E

typedef __attribute__((ext_vector_type(16))) _Float16 v16h;
typedef __attribute__((ext_vector_type(8)))  _Float16 v8h;
typedef __attribute__((ext_vector_type(8)))  float    v8f;

// ---------------------------------------------------------------------------
// Fragment builders from LDS rows laid out [outer][k] with 16B-aligned strides.
// A-matrix (16x32 f16): lanes 0-15 row M, K chunks {0..7,16..23}; lanes 16-31
// row M, K chunks {8..15,24..31}.
__device__ __forceinline__ v16h ldsFragA(const _Float16* rowBase, int lane) {
  const int k0 = (lane >> 4) << 3;               // 0 or 8
  v8h c0 = *(const v8h*)(rowBase + k0);
  v8h c1 = *(const v8h*)(rowBase + k0 + 16);
  v16h r;
#pragma unroll
  for (int j = 0; j < 8; ++j) { r[j] = c0[j]; r[j + 8] = c1[j]; }
  return r;
}
// B-matrix (32x16 f16): lanes 0-15 col N, K chunks {0..7,16..23}; lanes 16-31
// col N, K chunks {8..15,24..31}.
__device__ __forceinline__ v16h ldsFragB(const _Float16* rowBase, int lane) {
  const int k0 = (lane >> 4) << 3;               // 0 or 8
  v8h c0 = *(const v8h*)(rowBase + k0);
  v8h c1 = *(const v8h*)(rowBase + k0 + 16);
  v16h r;
#pragma unroll
  for (int j = 0; j < 8; ++j) { r[j] = c0[j]; r[j + 8] = c1[j]; }
  return r;
}

#define WMMA_F16(A, B, C) \
  __builtin_amdgcn_wmma_f32_16x16x32_f16(false, (A), false, (B), (short)0, (C), false, false)

// Complex tanh: tanh(a+ib) = (sinh 2a + i sin 2b) / (cosh 2a + cos 2b)
__device__ __forceinline__ void ctanhf(float a, float b, float& tr, float& ti) {
  float x = 2.0f * a, y = 2.0f * b;
  if (fabsf(x) > 20.0f) { tr = copysignf(1.0f, x); ti = 0.0f; return; }
  float sn, cs;
  __sincosf(y, &sn, &cs);
  float den = coshf(x) + cs;
  tr = sinhf(x) / den;
  ti = sn / den;
}

// ---------------------------------------------------------------------------
// Kernel 1: DFT bases, f16.
//   ffwdT[n][k], n=2m+c : c==0 -> cos(2pi m k / L), c==1 -> -sin(2pi m k / L)
//   finv [l][n], n=2m+c : c==0 -> a_m cos(2pi m l / L), c==1 -> -a_m sin(...)
//       a_0 = 1/L (imag of DC ignored by irfft), a_m = 2/L for m>=1
__global__ void __launch_bounds__(256)
basis_kernel(_Float16* __restrict__ ffwdT, _Float16* __restrict__ finv) {
  const float TWO_PI = 6.283185307179586f;
  int idx = blockIdx.x * 256 + threadIdx.x;
  if (idx < NC * LL) {
    int n = idx >> 11, kk = idx & (LL - 1);
    int m = n >> 1;
    int ph = (m * kk) & (LL - 1);                 // phase reduction for accuracy
    float s, c;
    __sincosf(TWO_PI * (float)ph * (1.0f / LL), &s, &c);
    ffwdT[idx] = (_Float16)((n & 1) ? -s : c);
  } else {
    int j = idx - NC * LL;
    if (j < LL * NC) {
      int l = j >> 7, n = j & (NC - 1);
      int m = n >> 1;
      int ph = (m * l) & (LL - 1);
      float s, c;
      __sincosf(TWO_PI * (float)ph * (1.0f / LL), &s, &c);
      float amp = (m == 0) ? (1.0f / LL) : (2.0f / LL);
      float v = (n & 1) ? ((m == 0) ? 0.0f : -amp * s) : amp * c;
      finv[j] = (_Float16)v;
    }
  }
}

// ---------------------------------------------------------------------------
// Kernel 2: forward DFT as GEMM with double-buffered LDS panels and 2x4
// register tiles per wave.  rows r=(b*H+h)*E+e from src[B,E,H,L]; out
// spec[r][2m+c] f32.  Block: 256 thr / 8 waves, tile 128 rows x 128 cols,
// K-panels of 32.  grid = (MROWS/128, 2) -- y selects q vs k.
__global__ void __launch_bounds__(256)
fwd_dft_kernel(const float* __restrict__ qg, const float* __restrict__ kg,
               const _Float16* __restrict__ fT,
               float* __restrict__ qspec, float* __restrict__ kspec) {
  const float* src = blockIdx.y ? kg : qg;
  float* dst       = blockIdx.y ? kspec : qspec;
  __shared__ __align__(16) _Float16 As[2][128][48];  // 48-halve stride = 96B
  __shared__ __align__(16) _Float16 Bs[2][128][48];  // 2*(12KB+12KB) = 48KB

  const int t = threadIdx.x, lane = t & 31, w = t >> 5;
  const int rowHalf = w >> 1;                    // 0..3 -> rows rowHalf*32 .. +31
  const int colBase = (w & 1) * 64;              // each wave: 4 col tiles
  const int blockRow = blockIdx.x * 128;

  // Per-thread A-panel load slots: 128 rows x 32 k = 1024 float4 / 256 thr = 4.
  const float* aptr[4];
  int arow[4], acol[4];
#pragma unroll
  for (int i = 0; i < 4; ++i) {
    int slot = t + i * 256;
    int row = slot >> 3, c4 = (slot & 7) * 4;
    int r = blockRow + row;
    int bi = r >> 9, hi = (r >> 6) & 7, ei = r & 63;   // BEHL -> (b,h,e) transpose
    aptr[i] = src + ((size_t)((bi * EE + ei) * HH + hi)) * LL + c4;
    arow[i] = row; acol[i] = c4;
  }
  // Per-thread B-panel slot: 128 n x 32 k halves / 256 thr = 16 halves each.
  const int bn = t >> 1, bk = (t & 1) * 16;
  const _Float16* bptr = fT + (size_t)bn * LL + bk;

  float4 ra[4];
  v8h rb0, rb1;

  // Prologue: fetch + commit panel k0=0 into buffer 0.
#pragma unroll
  for (int i = 0; i < 4; ++i) ra[i] = *(const float4*)(aptr[i]);
  rb0 = *(const v8h*)(bptr);
  rb1 = *(const v8h*)(bptr + 8);
#pragma unroll
  for (int i = 0; i < 4; ++i) {
    As[0][arow[i]][acol[i] + 0] = (_Float16)ra[i].x;
    As[0][arow[i]][acol[i] + 1] = (_Float16)ra[i].y;
    As[0][arow[i]][acol[i] + 2] = (_Float16)ra[i].z;
    As[0][arow[i]][acol[i] + 3] = (_Float16)ra[i].w;
  }
  *(v8h*)&Bs[0][bn][bk]     = rb0;
  *(v8h*)&Bs[0][bn][bk + 8] = rb1;
  __syncthreads();

  v8f acc[2][4] = {};
  int buf = 0;
  for (int k0 = 0; k0 < LL; k0 += 32) {
    const bool more = (k0 + 32) < LL;
    // Issue next panel's global loads first: latency overlaps the WMMA block.
    if (more) {
#pragma unroll
      for (int i = 0; i < 4; ++i) {
        ra[i] = *(const float4*)(aptr[i] + k0 + 32);
        if (i == 0 && k0 + 64 < LL)
          __builtin_prefetch(aptr[0] + k0 + 64, 0, 1);   // global_prefetch_b8
      }
      rb0 = *(const v8h*)(bptr + k0 + 32);
      rb1 = *(const v8h*)(bptr + k0 + 40);
    }
    // Compute from current buffer: 2 A frags + 4 B frags -> 8 WMMAs.
    v16h a0 = ldsFragA(&As[buf][rowHalf * 32 + (lane & 15)][0], lane);
    v16h a1 = ldsFragA(&As[buf][rowHalf * 32 + 16 + (lane & 15)][0], lane);
#pragma unroll
    for (int c = 0; c < 4; ++c) {
      v16h b = ldsFragB(&Bs[buf][colBase + c * 16 + (lane & 15)][0], lane);
      acc[0][c] = WMMA_F16(a0, b, acc[0][c]);
      acc[1][c] = WMMA_F16(a1, b, acc[1][c]);
    }
    // Commit next panel to the alternate buffer (its readers finished last iter).
    if (more) {
      int nb = buf ^ 1;
#pragma unroll
      for (int i = 0; i < 4; ++i) {
        As[nb][arow[i]][acol[i] + 0] = (_Float16)ra[i].x;
        As[nb][arow[i]][acol[i] + 1] = (_Float16)ra[i].y;
        As[nb][arow[i]][acol[i] + 2] = (_Float16)ra[i].z;
        As[nb][arow[i]][acol[i] + 3] = (_Float16)ra[i].w;
      }
      *(v8h*)&Bs[nb][bn][bk]     = rb0;
      *(v8h*)&Bs[nb][bn][bk + 8] = rb1;
    }
    __syncthreads();
    buf ^= 1;
  }

  const int rl = (lane >> 4) * 8;
#pragma unroll
  for (int rt = 0; rt < 2; ++rt)
#pragma unroll
    for (int c = 0; c < 4; ++c)
#pragma unroll
      for (int j = 0; j < 8; ++j) {
        int r = blockRow + rowHalf * 32 + rt * 16 + rl + j;
        int n = colBase + c * 16 + (lane & 15);
        dst[(size_t)r * NC + n] = acc[rt][c][j];
      }
}

// ---------------------------------------------------------------------------
// Kernel 3: per-(b,h) middle stage.  S = Q^T K (complex), tanh, T = K * tanh(S)^T.
// One block per bh, 256 thr / 8 waves.  LDS arena of 7 slots, stage-overlayed.
#define PS 72   // 72 halves = 144B stride (16B multiple)
__global__ void __launch_bounds__(256)
mid_kernel(const float* __restrict__ qspec, const float* __restrict__ kspec,
           float* __restrict__ tspec) {
  __shared__ __align__(16) _Float16 smem[7 * 64 * PS];
  _Float16* a0 = smem;                 // QreT        -> later KreN
  _Float16* a1 = smem + 1 * 64 * PS;   // -QimT       -> later -KimN
  _Float16* a2 = smem + 2 * 64 * PS;   // +QimT       -> later +KimN
  _Float16* a3 = smem + 3 * 64 * PS;   // KreT [y][e]
  _Float16* a4 = smem + 4 * 64 * PS;   // KimT [y][e]
  _Float16* a5 = smem + 5 * 64 * PS;   // tanh(S)re [x][y]
  _Float16* a6 = smem + 6 * 64 * PS;   // tanh(S)im [x][y]

  const int bh = blockIdx.x;
  const float* qb = qspec + (size_t)bh * EE * NC;
  const float* kb = kspec + (size_t)bh * EE * NC;
  const int t = threadIdx.x, lane = t & 31, w = t >> 5;

  // Load + transpose spectra into LDS (f32 -> f16), negation copies for complex GEMM.
  for (int idx = t; idx < EE * MODES; idx += 256) {
    int e = idx >> 6, x = idx & 63;
    float qr = qb[e * NC + 2 * x], qi = qb[e * NC + 2 * x + 1];
    float kr = kb[e * NC + 2 * x], ki = kb[e * NC + 2 * x + 1];
    a0[x * PS + e] = (_Float16)qr;
    a1[x * PS + e] = (_Float16)(-qi);
    a2[x * PS + e] = (_Float16)qi;
    a3[x * PS + e] = (_Float16)kr;
    a4[x * PS + e] = (_Float16)ki;
  }
  __syncthreads();

  // Stage 1: S[x][y] = sum_e Q[e][x]*K[e][y]  (complex, contraction over e)
#pragma unroll
  for (int rep = 0; rep < 2; ++rep) {
    int tid = w + rep * 8;
    int xt = tid >> 2, yt = tid & 3;
    v8f cre = {}, cim = {};
#pragma unroll
    for (int kb2 = 0; kb2 < EE; kb2 += 32) {
      v16h fre = ldsFragA(a0 + (xt * 16 + (lane & 15)) * PS + kb2, lane);
      v16h fin = ldsFragA(a1 + (xt * 16 + (lane & 15)) * PS + kb2, lane);
      v16h fip = ldsFragA(a2 + (xt * 16 + (lane & 15)) * PS + kb2, lane);
      v16h bre = ldsFragB(a3 + (yt * 16 + (lane & 15)) * PS + kb2, lane);
      v16h bim = ldsFragB(a4 + (yt * 16 + (lane & 15)) * PS + kb2, lane);
      cre = WMMA_F16(fre, bre, cre);
      cre = WMMA_F16(fin, bim, cre);   // - qim*kim
      cim = WMMA_F16(fre, bim, cim);
      cim = WMMA_F16(fip, bre, cim);   // + qim*kre
    }
    // complex tanh at writeback
#pragma unroll
    for (int j = 0; j < 8; ++j) {
      int x = xt * 16 + ((lane >> 4) * 8) + j;
      int y = yt * 16 + (lane & 15);
      float tr, ti;
      ctanhf(cre[j], cim[j], tr, ti);
      a5[x * PS + y] = (_Float16)tr;
      a6[x * PS + y] = (_Float16)ti;
    }
  }
  __syncthreads();

  // Repack K into natural [e][y] layout over the dead Q slots.
  for (int idx = t; idx < EE * MODES; idx += 256) {
    int e = idx >> 6, x = idx & 63;
    float kr = (float)a3[x * PS + e];
    float ki = (float)a4[x * PS + e];
    a0[e * PS + x] = (_Float16)kr;
    a1[e * PS + x] = (_Float16)(-ki);
    a2[e * PS + x] = (_Float16)ki;
  }
  __syncthreads();

  // Stage 2: T[e][x] = sum_y K[e][y] * A[x][y]  (contraction over y)
#pragma unroll
  for (int rep = 0; rep < 2; ++rep) {
    int tid = w + rep * 8;
    int et = tid >> 2, xt = tid & 3;
    v8f cre = {}, cim = {};
#pragma unroll
    for (int kb2 = 0; kb2 < MODES; kb2 += 32) {
      v16h kre = ldsFragA(a0 + (et * 16 + (lane & 15)) * PS + kb2, lane);
      v16h kin = ldsFragA(a1 + (et * 16 + (lane & 15)) * PS + kb2, lane);
      v16h kip = ldsFragA(a2 + (et * 16 + (lane & 15)) * PS + kb2, lane);
      v16h sre = ldsFragB(a5 + (xt * 16 + (lane & 15)) * PS + kb2, lane);
      v16h sim = ldsFragB(a6 + (xt * 16 + (lane & 15)) * PS + kb2, lane);
      cre = WMMA_F16(kre, sre, cre);
      cre = WMMA_F16(kin, sim, cre);
      cim = WMMA_F16(kre, sim, cim);
      cim = WMMA_F16(kip, sre, cim);
    }
#pragma unroll
    for (int j = 0; j < 8; ++j) {
      int e = et * 16 + ((lane >> 4) * 8) + j;
      int x = xt * 16 + (lane & 15);
      tspec[((size_t)bh * EE + e) * NC + 2 * x]     = cre[j];
      tspec[((size_t)bh * EE + e) * NC + 2 * x + 1] = cim[j];
    }
  }
}

// ---------------------------------------------------------------------------
// Kernel 4: channel mixing out[b,o] = sum_e T[b,e] * W[e,o] per (h, mode x).
// grid = H*MODES blocks, 128 thr / 4 waves (one 16x16 col tile each).
__global__ void __launch_bounds__(128)
mix_kernel(const float* __restrict__ tspec, const float* __restrict__ wre,
           const float* __restrict__ wim, float* __restrict__ ospec) {
  const int h = blockIdx.x >> 6, x = blockIdx.x & 63;
  __shared__ __align__(16) _Float16 Are[16 * PS], Ain[16 * PS], Aip[16 * PS];
  __shared__ __align__(16) _Float16 Bre[64 * PS], Bim[64 * PS];
  const int t = threadIdx.x, lane = t & 31, w = t >> 5;

  for (int idx = t; idx < BB * EE; idx += 128) {
    int b = idx >> 6, e = idx & 63;
    size_t off = ((size_t)(b * HH + h) * EE + e) * NC + 2 * x;
    float re = tspec[off], im = tspec[off + 1];
    Are[b * PS + e] = (_Float16)re;
    Ain[b * PS + e] = (_Float16)(-im);
    Aip[b * PS + e] = (_Float16)im;
  }
  for (int idx = t; idx < EE * EE; idx += 128) {
    int o = idx >> 6, e = idx & 63;
    size_t off = ((size_t)(h * EE + e) * EE + o) * MODES + x;
    Bre[o * PS + e] = (_Float16)wre[off];
    Bim[o * PS + e] = (_Float16)wim[off];
  }
  __syncthreads();

  v8f cre = {}, cim = {};
#pragma unroll
  for (int kb2 = 0; kb2 < EE; kb2 += 32) {
    v16h are = ldsFragA(Are + (lane & 15) * PS + kb2, lane);
    v16h ain = ldsFragA(Ain + (lane & 15) * PS + kb2, lane);
    v16h aip = ldsFragA(Aip + (lane & 15) * PS + kb2, lane);
    v16h bre = ldsFragB(Bre + (w * 16 + (lane & 15)) * PS + kb2, lane);
    v16h bim = ldsFragB(Bim + (w * 16 + (lane & 15)) * PS + kb2, lane);
    cre = WMMA_F16(are, bre, cre);
    cre = WMMA_F16(ain, bim, cre);
    cim = WMMA_F16(are, bim, cim);
    cim = WMMA_F16(aip, bre, cim);
  }
  const float sc = 1.0f / (float)(EE * EE);      // reference's out_ft / (E*E)
#pragma unroll
  for (int j = 0; j < 8; ++j) {
    int b = ((lane >> 4) * 8) + j;
    int o = w * 16 + (lane & 15);
    size_t off = ((size_t)(b * HH + h) * EE + o) * NC + 2 * x;
    ospec[off]     = cre[j] * sc;
    ospec[off + 1] = cim[j] * sc;
  }
}

// ---------------------------------------------------------------------------
// Kernel 5: inverse DFT GEMM [8192 x 2048] = [8192 x 128] x [128 x 2048],
// scattering into [B,E,H,L] output. grid = (MROWS/64, LL/128), 256 thr.
__global__ void __launch_bounds__(256)
inv_dft_kernel(const float* __restrict__ ospec, const _Float16* __restrict__ finv,
               float* __restrict__ out) {
  __shared__ __align__(16) _Float16 As[64 * 144];   // 144 halves = 288B stride
  __shared__ __align__(16) _Float16 Bs[128 * 144];
  const int t = threadIdx.x, lane = t & 31, w = t >> 5;
  const int blockRow = blockIdx.x * 64;
  const int colBlock = blockIdx.y * 128;

  for (int idx = t; idx < 64 * NC; idx += 256) {
    int row = idx >> 7, c = idx & 127;
    As[row * 144 + c] = (_Float16)ospec[((size_t)(blockRow + row)) * NC + c];
  }
  for (int idx = t; idx < 2048; idx += 256) {       // 128 rows x 128 halves / 8
    int n = idx >> 4, kk = (idx & 15) * 8;
    *(v8h*)&Bs[n * 144 + kk] = *(const v8h*)(finv + ((size_t)(colBlock + n)) * NC + kk);
  }
  __syncthreads();

  const int rowTile = w & 3;
  const int colBase = (w >> 2) * 64;
  v8f acc[4] = {};
#pragma unroll
  for (int k0 = 0; k0 < NC; k0 += 32) {
    v16h a = ldsFragA(As + (rowTile * 16 + (lane & 15)) * 144 + k0, lane);
#pragma unroll
    for (int c = 0; c < 4; ++c) {
      v16h b = ldsFragB(Bs + (colBase + c * 16 + (lane & 15)) * 144 + k0, lane);
      acc[c] = WMMA_F16(a, b, acc[c]);
    }
  }

  const int rl = (lane >> 4) * 8;
#pragma unroll
  for (int c = 0; c < 4; ++c)
#pragma unroll
    for (int j = 0; j < 8; ++j) {
      int r = blockRow + rowTile * 16 + rl + j;
      int bi = r >> 9, hi = (r >> 6) & 7, oi = r & 63;
      int l = colBlock + colBase + c * 16 + (lane & 15);
      out[(((size_t)bi * EE + oi) * HH + hi) * LL + l] = acc[c][j];
    }
}

// ---------------------------------------------------------------------------
extern "C" void kernel_launch(void* const* d_in, const int* in_sizes, int n_in,
                              void* d_out, int out_size, void* d_ws, size_t ws_size,
                              hipStream_t stream) {
  (void)in_sizes; (void)n_in; (void)out_size; (void)ws_size;
  const float* q   = (const float*)d_in[0];
  const float* k   = (const float*)d_in[1];
  // d_in[2] = v : unused by the reference module
  const float* wre = (const float*)d_in[3];
  const float* wim = (const float*)d_in[4];
  float* out = (float*)d_out;

  // Workspace carve-up: 1 MB of f16 bases + 4x 4MB f32 spectra = 17 MB.
  char* ws = (char*)d_ws;
  _Float16* ffwdT = (_Float16*)ws;                       // [128][2048]
  _Float16* finv  = ffwdT + (size_t)NC * LL;             // [2048][128]
  float* qspec = (float*)(ws + 2 * (size_t)NC * LL * sizeof(_Float16));
  float* kspec = qspec + (size_t)MROWS * NC;
  float* tspec = kspec + (size_t)MROWS * NC;
  float* ospec = tspec + (size_t)MROWS * NC;

  basis_kernel<<<dim3((NC * LL + LL * NC + 255) / 256), dim3(256), 0, stream>>>(ffwdT, finv);
  fwd_dft_kernel<<<dim3(MROWS / 128, 2), dim3(256), 0, stream>>>(q, k, ffwdT, qspec, kspec);
  mid_kernel<<<dim3(BB * HH), dim3(256), 0, stream>>>(qspec, kspec, tspec);
  mix_kernel<<<dim3(HH * MODES), dim3(128), 0, stream>>>(tspec, wre, wim, ospec);
  inv_dft_kernel<<<dim3(MROWS / 64, LL / 128), dim3(256), 0, stream>>>(ospec, finv, out);
}
